// TranspConvBlock_27230092657172
// MI455X (gfx1250) — compile-verified, matrix operand order checked by reading, not run on hardware
//
#include <hip/hip_runtime.h>
#include <hip/hip_bf16.h>
#include <math.h>

// ---------------------------------------------------------------------------
// Problem constants
// ---------------------------------------------------------------------------
#define Bn   4
#define CIN  256
#define COUT 128
#define Hh   64
#define Ww   64
#define Hp   66          // padded (1-px zero halo)
#define Wp   66
#define P1   (Bn*Hh*Ww)  // 16384 pixels (stage 1/2)
#define H2   128
#define W2   128
#define P2   (Bn*H2*W2)  // 65536 pixels (stage 5)
#define EPSF 1e-5f

#define XHP_N   (Bn*Hp*Wp*CIN)   // padded x, f16 NHWC
#define Y1HP_N  (Bn*Hp*Wp*COUT)  // padded y1, f16 NHWC

typedef __attribute__((ext_vector_type(16))) _Float16 v16h;
typedef __attribute__((ext_vector_type(8)))  _Float16 v8h;
typedef __attribute__((ext_vector_type(8)))  float    v8f;

// ---------------------------------------------------------------------------
// WMMA fragment helpers (gfx1250 16-bit operand layout:
//   lanes 0-15  : K in {0..7} u {16..23}
//   lanes 16-31 : K in {8..15} u {24..31}
// so a fragment is two contiguous 8-half (16B) runs per lane)
// ---------------------------------------------------------------------------
__device__ __forceinline__ v16h load_frag(const _Float16* r0, const _Float16* r1) {
  v8h a = *(const v8h*)r0;
  v8h b = *(const v8h*)r1;
  v16h r;
#pragma unroll
  for (int j = 0; j < 8; ++j) { r[j] = a[j]; r[j + 8] = b[j]; }
  return r;
}
__device__ __forceinline__ v8f wmma16(v16h a, v16h b, v8f c) {
  return __builtin_amdgcn_wmma_f32_16x16x32_f16(false, a, false, b, (short)0, c,
                                                false, false);
}

// ---------------------------------------------------------------------------
// Stage 0 kernels
// ---------------------------------------------------------------------------
__global__ void zeroStats(float* stats) {
  int i = threadIdx.x;
  if (i < 1024) stats[i] = 0.f;
}

// x NCHW f32 -> xhp NHWC f16 with 1-px zero halo (66x66)
__global__ void xToHalf(const float* __restrict__ x, _Float16* __restrict__ xhp) {
  int idx = blockIdx.x * blockDim.x + threadIdx.x;
  if (idx >= XHP_N) return;
  int c = idx & 255;
  int p = idx >> 8;
  int xx = p % Wp;
  int yy = (p / Wp) % Hp;
  int b = p / (Hp * Wp);
  float v = 0.f;
  if (yy >= 1 && yy <= Hh && xx >= 1 && xx <= Ww)
    v = x[((b * CIN + c) * Hh + (yy - 1)) * Ww + (xx - 1)];
  xhp[idx] = (_Float16)v;
}

// off_w (27,256,3,3) -> offWp [n=32][k=tap*256+ci] f16, zero padded n>=27
__global__ void packOffW(const float* __restrict__ w, _Float16* __restrict__ wp) {
  int idx = blockIdx.x * blockDim.x + threadIdx.x;
  if (idx >= 32 * 2304) return;
  int n = idx / 2304, k = idx % 2304;
  int tap = k >> 8, ci = k & 255;
  float v = (n < 27) ? w[(n * CIN + ci) * 9 + tap] : 0.f;
  wp[idx] = (_Float16)v;
}

// dcn_w (128,256,3,3) -> dcnWp [n=128][k=tap*256+ci] f16
__global__ void packDcnW(const float* __restrict__ w, _Float16* __restrict__ wp) {
  int idx = blockIdx.x * blockDim.x + threadIdx.x;
  if (idx >= COUT * 2304) return;
  int n = idx / 2304, k = idx % 2304;
  int tap = k >> 8, ci = k & 255;
  wp[idx] = (_Float16)w[(n * CIN + ci) * 9 + tap];
}

// up_w (128,128,4,4) -> upWp [cls=4][n=128][k=s*128+ci] f16
// value = wf[co=n][ci][ky][kx] = up_w[ci][n][3-ky][3-kx],
// ky = py + 2*sy, kx = px + 2*sx, cls = py*2+px, s = sy*2+sx
__global__ void packUpW(const float* __restrict__ w, _Float16* __restrict__ wp) {
  int idx = blockIdx.x * blockDim.x + threadIdx.x;
  if (idx >= 4 * COUT * 512) return;
  int k   = idx & 511;
  int n   = (idx >> 9) & 127;
  int cls = idx >> 16;
  int s = k >> 7, ci = k & 127;
  int py = cls >> 1, px = cls & 1;
  int sy = s >> 1,  sx = s & 1;
  int ky = py + 2 * sy, kx = px + 2 * sx;
  wp[idx] = (_Float16)w[((ci * COUT + n) * 4 + (3 - ky)) * 4 + (3 - kx)];
}

// ---------------------------------------------------------------------------
// Stage 1: offset conv (3x3, 256 -> 27) as WMMA GEMM M=16384 N=32 K=2304.
// 128 threads = 4 waves, each wave owns one 16-pixel M tile, N as two
// 16-wide tiles.  Halo'd input: every tap load is unconditionally valid ->
// inner loop is pure load clauses + 16 WMMAs per tap, zero predicates.
// ---------------------------------------------------------------------------
__global__ __launch_bounds__(128) void offsetConvKernel(
    const _Float16* __restrict__ xhp, const _Float16* __restrict__ wp,
    const float* __restrict__ offB, float* __restrict__ off) {
  int lane = threadIdx.x & 31, wave = threadIdx.x >> 5;
  int pixBase = (blockIdx.x * 4 + wave) * 16;
  int b = pixBase >> 12, rem = pixBase & 4095, y = rem >> 6, x0 = rem & 63;
  int kShift = (lane >> 4) * 8;
  int col = lane & 15;
  int xc = x0 + col;

  const _Float16* wp0 = wp + col * 2304 + kShift;
  const _Float16* wp1 = wp + (col + 16) * 2304 + kShift;

  v8f acc0 = {}, acc1 = {};
#pragma unroll 1
  for (int ky = 0; ky < 3; ++ky) {
#pragma unroll 1
    for (int kx = 0; kx < 3; ++kx) {
      // padded index: (y + ky, xc + kx) == (yy+1, xx+1), always in-bounds
      const _Float16* ap =
          xhp + (((b * Hp + (y + ky)) * Wp + (xc + kx)) * CIN + kShift);
      int tapK = (ky * 3 + kx) * 256;
#pragma unroll
      for (int kc = 0; kc < 8; ++kc) {
        int cc = kc * 32;
        v16h a = load_frag(ap + cc, ap + cc + 16);
        acc0 = wmma16(a, load_frag(wp0 + tapK + cc, wp0 + tapK + cc + 16), acc0);
        acc1 = wmma16(a, load_frag(wp1 + tapK + cc, wp1 + tapK + cc + 16), acc1);
      }
    }
  }

#pragma unroll
  for (int nt = 0; nt < 2; ++nt) {
    int n = nt * 16 + col;
    if (n >= 27) continue;
    float bias = offB[n];
    v8f acc = nt ? acc1 : acc0;
#pragma unroll
    for (int v = 0; v < 8; ++v) {
      int m = v + 8 * (lane >> 4);
      float val = acc[v] + bias;
      if (n >= 18) val = 1.f / (1.f + __expf(-val));  // sigmoid(mask)
      off[(pixBase + m) * 32 + n] = val;
    }
  }
}

// ---------------------------------------------------------------------------
// Stage 2: deformable conv GEMM M=16384 N=128 K=2304.
// 256 threads = 8 waves; block owns a 16-pixel tile.  Per tap: cooperative
// bilinear gather of a 16x256 f16 A-panel into LDS (mask folded in).
// Neighbor indices clamp into the zero halo, so out-of-bounds taps read
// zeros with no validity arithmetic.  Each wave then drives one 16-wide
// N tile through 8 WMMAs against dcnWp.
// ---------------------------------------------------------------------------
__global__ __launch_bounds__(256) void deformGemm(
    const _Float16* __restrict__ xhp, const _Float16* __restrict__ wp,
    const float* __restrict__ off, const float* __restrict__ dcnB,
    float* __restrict__ y0out) {
  __shared__ _Float16 aT[16][264];  // 16 pixels x 256 K (+8 pad)
  int tid = threadIdx.x;
  int lane = tid & 31, wave = tid >> 5;
  int pixBase = blockIdx.x * 16;

  // fill-phase coordinates: thread handles pixel tid/16, channels (tid%16)*16..
  int pf = pixBase + (tid >> 4);
  int bF = pf >> 12, remF = pf & 4095, yF = remF >> 6, xF = remF & 63;
  int cbase = (tid & 15) * 16;

  int rowW = lane & 15;
  int kShift = (lane >> 4) * 8;
  int nCh = wave * 16 + rowW;
  const _Float16* wpn = wp + nCh * 2304 + kShift;

  v8f acc = {};
#pragma unroll 1
  for (int tap = 0; tap < 9; ++tap) {
    // --- bilinear sample into LDS -------------------------------------
    float dy = off[pf * 32 + tap];
    float dx = off[pf * 32 + 9 + tap];
    float mk = off[pf * 32 + 18 + tap];
    float py = (float)yF + (float)(tap / 3 - 1) + dy;
    float px = (float)xF + (float)(tap % 3 - 1) + dx;
    float fy = floorf(py), fx = floorf(px);
    int y0i = (int)fy, x0i = (int)fx;
    float wy = py - fy, wx = px - fx;

    float wgt[4];
    v8h nb[4][2];
#pragma unroll
    for (int q = 0; q < 4; ++q) {
      // clamp into [-1,64]: out-of-range taps land on the zero halo
      int yi = min(max(y0i + (q >> 1), -1), Hh);
      int xi = min(max(x0i + (q & 1), -1), Ww);
      float wq = ((q >> 1) ? wy : (1.f - wy)) * ((q & 1) ? wx : (1.f - wx));
      wgt[q] = wq * mk;
      const _Float16* p =
          xhp + ((bF * Hp + (yi + 1)) * Wp + (xi + 1)) * CIN + cbase;
      nb[q][0] = *(const v8h*)p;
      nb[q][1] = *(const v8h*)(p + 8);
    }
    __syncthreads();  // protect previous tap's LDS reads
    v8h o0, o1;
#pragma unroll
    for (int j = 0; j < 8; ++j) {
      o0[j] = (_Float16)(wgt[0] * (float)nb[0][0][j] + wgt[1] * (float)nb[1][0][j] +
                         wgt[2] * (float)nb[2][0][j] + wgt[3] * (float)nb[3][0][j]);
      o1[j] = (_Float16)(wgt[0] * (float)nb[0][1][j] + wgt[1] * (float)nb[1][1][j] +
                         wgt[2] * (float)nb[2][1][j] + wgt[3] * (float)nb[3][1][j]);
    }
    *(v8h*)&aT[tid >> 4][cbase] = o0;
    *(v8h*)&aT[tid >> 4][cbase + 8] = o1;
    __syncthreads();

    // --- WMMA over this tap's 256-wide K panel ------------------------
    const _Float16* wt = wpn + tap * 256;
#pragma unroll
    for (int kc = 0; kc < 8; ++kc) {
      int kb = kc * 32 + kShift;
      v16h a = load_frag(&aT[rowW][kb], &aT[rowW][kb + 16]);
      acc = wmma16(a, load_frag(wt + kc * 32, wt + kc * 32 + 16), acc);
    }
  }

  float bias = dcnB[nCh];
#pragma unroll
  for (int v = 0; v < 8; ++v) {
    int m = v + 8 * (lane >> 4);
    y0out[(pixBase + m) * COUT + nCh] = acc[v] + bias;
  }
}

// ---------------------------------------------------------------------------
// Per-channel stats: partial sums with f32 global atomics (channel = tid%128)
// ---------------------------------------------------------------------------
__global__ void reduceStats(const float* __restrict__ y, int npix,
                            float* __restrict__ sum, float* __restrict__ sq,
                            int slots) {
  int c = threadIdx.x & 127;
  int slot = blockIdx.x * (blockDim.x >> 7) + (threadIdx.x >> 7);
  float s = 0.f, s2 = 0.f;
  for (int p = slot; p < npix; p += slots) {
    float v = y[p * COUT + c];
    s += v;
    s2 += v * v;
  }
  atomicAdd(&sum[c], s);
  atomicAdd(&sq[c], s2);
}

__global__ void bnParams(const float* __restrict__ sum, const float* __restrict__ sq,
                         const float* __restrict__ g, const float* __restrict__ bb,
                         float invN, float* __restrict__ a, float* __restrict__ c) {
  int ch = threadIdx.x;
  if (ch >= COUT) return;
  float mu = sum[ch] * invN;
  float var = sq[ch] * invN - mu * mu;
  float s = g[ch] * rsqrtf(var + EPSF);
  a[ch] = s;
  c[ch] = bb[ch] - mu * s;
}

// BN+ReLU, f32 NHWC -> f16 NHWC with 1-px zero halo (66x66)
__global__ void makeY1h(const float* __restrict__ y0, const float* __restrict__ a,
                        const float* __restrict__ c, _Float16* __restrict__ y1hp) {
  int idx = blockIdx.x * blockDim.x + threadIdx.x;
  if (idx >= Y1HP_N) return;
  int ch = idx & 127;
  int p = idx >> 7;
  int xx = p % Wp;
  int yy = (p / Wp) % Hp;
  int b = p / (Hp * Wp);
  float v = 0.f;
  if (yy >= 1 && yy <= Hh && xx >= 1 && xx <= Ww) {
    float t = y0[((b * Hh + (yy - 1)) * Ww + (xx - 1)) * COUT + ch] * a[ch] + c[ch];
    v = fmaxf(t, 0.f);
  }
  y1hp[idx] = (_Float16)v;
}

// ---------------------------------------------------------------------------
// Stage 5: stride-2 transpose conv as 4 parity-class GEMMs, each
// M=16384 N=128 K=512 (2x2 taps x 128 ci).  128 threads = 4 waves, wave
// handles two 16-wide N tiles; A fragments loaded straight from halo'd
// y1hp (always valid).  K loop fully unrolled -> 32 unconditional WMMAs.
// ---------------------------------------------------------------------------
__global__ __launch_bounds__(128) void upGemm(const _Float16* __restrict__ y1hp,
                                              const _Float16* __restrict__ wp,
                                              float* __restrict__ z) {
  int lane = threadIdx.x & 31, wave = threadIdx.x >> 5;
  int cls = blockIdx.x >> 10;   // 4 classes x 1024 M tiles
  int mt = blockIdx.x & 1023;
  int pixBase = mt * 16;
  int py = cls >> 1, px = cls & 1;

  int p = pixBase + (lane & 15);
  int b = p >> 12, rem = p & 4095, ty = rem >> 6, tx = rem & 63;
  int kShift = (lane >> 4) * 8;
  int col = lane & 15;

  int n0 = (wave * 2 + 0) * 16 + col;
  int n1 = (wave * 2 + 1) * 16 + col;
  const _Float16* wb0 = wp + (cls * COUT + n0) * 512 + kShift;
  const _Float16* wb1 = wp + (cls * COUT + n1) * 512 + kShift;

  v8f acc0 = {}, acc1 = {};
#pragma unroll
  for (int kc = 0; kc < 16; ++kc) {
    const int seg = kc >> 2, sy = seg >> 1, sx = seg & 1;
    // padded index: (ty+py+sy, tx+px+sx) in [0,65], always valid
    const _Float16* ap =
        y1hp + (((b * Hp + (ty + py + sy)) * Wp + (tx + px + sx)) * COUT +
                (kc & 3) * 32 + kShift);
    v16h a = load_frag(ap, ap + 16);
    int kb = kc * 32;
    acc0 = wmma16(a, load_frag(wb0 + kb, wb0 + kb + 16), acc0);
    acc1 = wmma16(a, load_frag(wb1 + kb, wb1 + kb + 16), acc1);
  }

#pragma unroll
  for (int t = 0; t < 2; ++t) {
    int n = (wave * 2 + t) * 16 + col;
    v8f acc = t ? acc1 : acc0;
#pragma unroll
    for (int v = 0; v < 8; ++v) {
      int m = v + 8 * (lane >> 4);
      int pp = pixBase + m;
      int bb = pp >> 12, rr = pp & 4095, tyy = rr >> 6, txx = rr & 63;
      int oy = 2 * tyy + py, ox = 2 * txx + px;
      z[((bb * H2 + oy) * W2 + ox) * COUT + n] = acc[v];
    }
  }
}

// BN+ReLU, NHWC f32 -> NCHW f32 output
__global__ void finalOut(const float* __restrict__ z, const float* __restrict__ a,
                         const float* __restrict__ c, float* __restrict__ out) {
  int idx = blockIdx.x * blockDim.x + threadIdx.x;
  if (idx >= P2 * COUT) return;
  int ox = idx & 127;
  int oy = (idx >> 7) & 127;
  int ch = (idx >> 14) & 127;
  int b = idx >> 21;
  float v = z[((b * H2 + oy) * W2 + ox) * COUT + ch] * a[ch] + c[ch];
  out[idx] = fmaxf(v, 0.f);
}

// ---------------------------------------------------------------------------
// Launch
// ---------------------------------------------------------------------------
extern "C" void kernel_launch(void* const* d_in, const int* in_sizes, int n_in,
                              void* d_out, int out_size, void* d_ws, size_t ws_size,
                              hipStream_t stream) {
  const float* x     = (const float*)d_in[0];
  const float* off_w = (const float*)d_in[1];
  const float* off_b = (const float*)d_in[2];
  const float* dcn_w = (const float*)d_in[3];
  const float* dcn_b = (const float*)d_in[4];
  const float* bn1_g = (const float*)d_in[5];
  const float* bn1_b = (const float*)d_in[6];
  const float* up_w  = (const float*)d_in[7];
  const float* bn2_g = (const float*)d_in[8];
  const float* bn2_b = (const float*)d_in[9];
  float* out = (float*)d_out;

  char* base = (char*)d_ws;
  size_t o = 0;
  _Float16* xhp   = (_Float16*)(base + o); o += (size_t)XHP_N * 2;      // 8.9 MB
  _Float16* offWp = (_Float16*)(base + o); o += (size_t)32 * 2304 * 2;
  _Float16* dcnWp = (_Float16*)(base + o); o += (size_t)COUT * 2304 * 2;
  _Float16* upWp  = (_Float16*)(base + o); o += (size_t)4 * COUT * 512 * 2;
  float*    off   = (float*)(base + o);    o += (size_t)P1 * 32 * 4;    // 2.0 MB
  float*    y0    = (float*)(base + o);    o += (size_t)P1 * COUT * 4;  // 8.0 MB
  _Float16* y1hp  = (_Float16*)(base + o); o += (size_t)Y1HP_N * 2;     // 4.5 MB
  float*    z     = (float*)(base + o);    o += (size_t)P2 * COUT * 4;  // 32 MB
  float*    stats = (float*)(base + o);    o += 1024 * 4;
  float* sum1 = stats + 0,   *sq1 = stats + 128;
  float* a1   = stats + 256, *c1  = stats + 384;
  float* sum2 = stats + 512, *sq2 = stats + 640;
  float* a2   = stats + 768, *c2  = stats + 896;

  zeroStats<<<1, 1024, 0, stream>>>(stats);
  xToHalf<<<(XHP_N + 511) / 512, 512, 0, stream>>>(x, xhp);
  packOffW<<<(32 * 2304 + 255) / 256, 256, 0, stream>>>(off_w, offWp);
  packDcnW<<<(COUT * 2304 + 255) / 256, 256, 0, stream>>>(dcn_w, dcnWp);
  packUpW<<<(4 * COUT * 512 + 255) / 256, 256, 0, stream>>>(up_w, upWp);

  offsetConvKernel<<<P1 / 64, 128, 0, stream>>>(xhp, offWp, off_b, off);
  deformGemm<<<P1 / 16, 256, 0, stream>>>(xhp, dcnWp, off, dcn_b, y0);

  reduceStats<<<128, 256, 0, stream>>>(y0, P1, sum1, sq1, 256);
  bnParams<<<1, 128, 0, stream>>>(sum1, sq1, bn1_g, bn1_b, 1.f / (float)P1, a1, c1);
  makeY1h<<<(Y1HP_N + 255) / 256, 256, 0, stream>>>(y0, a1, c1, y1hp);

  upGemm<<<4 * (P1 / 16), 128, 0, stream>>>(y1hp, upWp, z);

  reduceStats<<<256, 256, 0, stream>>>(z, P2, sum2, sq2, 512);
  bnParams<<<1, 128, 0, stream>>>(sum2, sq2, bn2_g, bn2_b, 1.f / (float)P2, a2, c2);
  finalOut<<<(P2 * COUT + 255) / 256, 256, 0, stream>>>(z, a2, c2, out);
}
